// MultiHeadAttention_24326694764567
// MI455X (gfx1250) — matrix operand order repaired and checked by reading, not compiled, not run on hardware
//
#include <hip/hip_runtime.h>
#include <stdint.h>

typedef _Float16 h16;
typedef __attribute__((ext_vector_type(16))) _Float16 v16h;
typedef __attribute__((ext_vector_type(8)))  float    v8f;
typedef __attribute__((ext_vector_type(4)))  uint32_t u32x4;
typedef __attribute__((ext_vector_type(8)))  uint32_t u32x8;

#define D_MODEL 1024
#define T_LEN   2048
#define BATCH   2
#define NHEAD   16
#define HDIM    64
#define M_TOT   (BATCH * T_LEN)   // 4096 tokens total

// ---------------------------------------------------------------------------
// WMMA fragment loaders (CDNA5 16-bit layouts, wave32)
// A (16x32, MxK): lane L holds row M=L%16; VGPR r holds K pair
//   K = (r/4)*16 + (L/16)*8 + (r%4)*2   (+0,+1)
// B (32x16, KxN), source stored transposed [N][K]: lane L holds col N=L%16;
//   VGPR r holds K pair K = (r/4)*16 + (L/16)*8 + (r%4)*2 (same as A).
// C/D f32 (16x16): lane L, VGPR r -> M = (L/16)*8 + r, N = L%16.
// ---------------------------------------------------------------------------
__device__ __forceinline__ v16h frag_a_rowmajor(const h16* base, int ld, int lane) {
    int m = lane & 15, g = lane >> 4;
    union { v16h v; uint32_t u[8]; } f;
    const uint32_t* row = (const uint32_t*)(base + (size_t)m * ld);
    #pragma unroll
    for (int r = 0; r < 8; ++r) {
        int k = ((r >> 2) << 4) + (g << 3) + ((r & 3) << 1);
        f.u[r] = row[k >> 1];
    }
    return f.v;
}

__device__ __forceinline__ v16h frag_b_nk(const h16* baseT, int ld, int lane) {
    int n = lane & 15, g = lane >> 4;
    union { v16h v; uint32_t u[8]; } f;
    const uint32_t* row = (const uint32_t*)(baseT + (size_t)n * ld);
    #pragma unroll
    for (int r = 0; r < 8; ++r) f.u[r] = row[((r >> 2) << 3) + (g << 2) + (r & 3)];   // two 16-byte runs
    return f.v;
}

// ---------------------------------------------------------------------------
// Tensor Data Mover: DMA a 2-D f16 tile (tile_d1 rows x tile_d0 elems, row
// stride stride0 elems) from global to LDS. Descriptor per CDNA5 ISA:
//  group0: count=1 | lds_addr | global_addr(57b) | type=2
//  group1: data_size=2B, tensor dims (large -> never OOB-dropped), tile dims,
//          dim0 stride
//  group2/3: tile_dim2/3/4 = 0 (unused), tensor dims nonzero.
// All operands are wave-uniform (built from blockIdx) -> SGPRs.
// ---------------------------------------------------------------------------
__device__ __forceinline__ void tdm_load_tile_f16(uint32_t lds_off, const h16* gptr,
                                                  uint32_t tile_d0, uint32_t tile_d1,
                                                  uint32_t stride0) {
    uint64_t ga = (uint64_t)(uintptr_t)gptr;
    u32x4 g0 = { 1u,                                   // count=1 valid descriptor
                 lds_off,                              // LDS byte address
                 (uint32_t)ga,                         // global addr [31:0]
                 ((uint32_t)(ga >> 32) & 0x01FFFFFFu) | (2u << 30) }; // [56:32]|type=2
    const uint32_t td0 = 1u << 20, td1 = 1u << 16;     // generous tensor dims
    u32x8 g1;
    g1[0] = 0x00010000u;                               // wg_mask=0, data_size=2B
    g1[1] = (td0 & 0xFFFFu) << 16;                     // tensor_dim0 lo
    g1[2] = (td0 >> 16) | ((td1 & 0xFFFFu) << 16);     // tensor_dim0 hi | dim1 lo
    g1[3] = (td1 >> 16) | (tile_d0 << 16);             // tensor_dim1 hi | tile_dim0
    g1[4] = tile_d1;                                   // tile_dim1, tile_dim2=0
    g1[5] = stride0;                                   // tensor_dim0_stride lo
    g1[6] = 0;                                         // stride0 hi | dim1_stride lo
    g1[7] = 0;
    u32x4 g2 = { 1u << 20, 1u << 20, 0u, 0u };         // tensor_dim2/3 >0, tile_dim3=0
    u32x4 g3 = { 0u, 0xFFFF0000u, 0u, 0u };            // tensor_dim4=0xFFFF, tile_dim4=0
    asm volatile("tensor_load_to_lds %0, %1, %2, %3"
                 :: "s"(g0), "s"(g1), "s"(g2), "s"(g3)
                 : "memory");
}

// ---------------------------------------------------------------------------
// Stage 0: casts
// ---------------------------------------------------------------------------
__global__ __launch_bounds__(256) void cast_f16_kernel(const float* __restrict__ src,
                                                       h16* __restrict__ dst, int n) {
    int i = blockIdx.x * 256 + threadIdx.x;
    if (i < n) dst[i] = (h16)src[i];
}

// W [K][N] f32 (1024x1024) -> Wt [N][K] f16
__global__ __launch_bounds__(256) void transpose_cast_kernel(const float* __restrict__ W,
                                                             h16* __restrict__ Wt) {
    int i = blockIdx.x * 256 + threadIdx.x;     // over 1M elements
    int k = i >> 10, n = i & 1023;
    Wt[((size_t)n << 10) + k] = (h16)W[i];
}

// ---------------------------------------------------------------------------
// Stage 1: QKV projection GEMM.  X[4096x1024] f16 @ Wt[N][K] f16.
// One wave -> 32(M)x64(N) tile (two A frags share each B frag: 8 WMMA / 12
// b128 loads per k-step). Block = 8 waves -> 256x64; grid (16, 16).
// transposed==0: dst[b][h][t][d] ; transposed==1: dst[b][h][d][t]
// ---------------------------------------------------------------------------
__global__ __launch_bounds__(256) void gemm_qkv_kernel(const h16* __restrict__ X,
                                                       const h16* __restrict__ Wt,
                                                       h16* __restrict__ dst,
                                                       int transposed) {
    int lane = threadIdx.x & 31, wid = threadIdx.x >> 5;
    int m0 = blockIdx.y * 256 + wid * 32;
    int n0 = blockIdx.x * 64;
    v8f zero = {};
    v8f acc[2][4] = {{zero, zero, zero, zero}, {zero, zero, zero, zero}};
    for (int kk = 0; kk < D_MODEL; kk += 32) {
        v16h a0 = frag_a_rowmajor(X + (size_t)m0 * D_MODEL + kk, D_MODEL, lane);
        v16h a1 = frag_a_rowmajor(X + (size_t)(m0 + 16) * D_MODEL + kk, D_MODEL, lane);
        #pragma unroll
        for (int nt = 0; nt < 4; ++nt) {
            v16h bm = frag_b_nk(Wt + (size_t)(n0 + nt * 16) * D_MODEL + kk, D_MODEL, lane);
            acc[0][nt] = __builtin_amdgcn_wmma_f32_16x16x32_f16(false, a0, false, bm,
                                                                (short)0, acc[0][nt], false, false);
            acc[1][nt] = __builtin_amdgcn_wmma_f32_16x16x32_f16(false, a1, false, bm,
                                                                (short)0, acc[1][nt], false, false);
        }
    }
    int g = lane >> 4, nl = lane & 15;
    #pragma unroll
    for (int s = 0; s < 2; ++s) {
        #pragma unroll
        for (int nt = 0; nt < 4; ++nt) {
            #pragma unroll
            for (int r = 0; r < 8; ++r) {
                int rowg = m0 + s * 16 + (g << 3) + r;     // global token row
                int b = rowg >> 11, t = rowg & (T_LEN - 1);
                int col = n0 + nt * 16 + nl;
                int h = col >> 6, d = col & (HDIM - 1);
                h16 v = (h16)acc[s][nt][r];
                if (transposed)
                    dst[(((size_t)(b * NHEAD + h)) * HDIM + d) * T_LEN + t] = v;
                else
                    dst[(((size_t)(b * NHEAD + h)) * T_LEN + t) * HDIM + d] = v;
            }
        }
    }
}

// ---------------------------------------------------------------------------
// Stage 2: causal flash attention. ONE WAVE per workgroup so that all tile
// addresses are scalar (blockIdx-derived) -> TDM descriptors live in SGPRs.
// K/V tiles are DMA'd by the Tensor Data Mover into double-buffered LDS,
// prefetching block jb+32 while computing block jb (TENSORcnt-synchronized).
// Q,K: [bh][t][64] f16 ; V: [bh][64][t] f16 ; ctx out: [b][t][1024] f16.
// ---------------------------------------------------------------------------
__global__ __launch_bounds__(32) void attn_fa_kernel(const h16* __restrict__ Q,
                                                     const h16* __restrict__ K,
                                                     const h16* __restrict__ Vt,
                                                     h16* __restrict__ ctx) {
    __shared__ h16 kbuf[2][32 * HDIM];   // 2 x 4KB : 32 keys x 64 d (row-major)
    __shared__ h16 vbuf[2][HDIM * 32];   // 2 x 4KB : 64 d x 32 keys (row-major)
    __shared__ h16 pbuf[16 * 32];        // 1KB     : P tile
    int lane = threadIdx.x;
    int qtile = blockIdx.x & 127;        // uniform (scalar)
    int bh = blockIdx.x >> 7;            // uniform (scalar)
    int b = bh >> 4, h = bh & 15;
    int g = lane >> 4, nl = lane & 15;
    int qbase = qtile << 4;
    int qlimit = qbase + 16;

    const h16* Qp = Q  + (size_t)bh * T_LEN * HDIM;
    const h16* Kp = K  + (size_t)bh * T_LEN * HDIM;
    const h16* Vp = Vt + (size_t)bh * HDIM * T_LEN;

    v16h qf0 = frag_a_rowmajor(Qp + (size_t)qbase * HDIM,      HDIM, lane);
    v16h qf1 = frag_a_rowmajor(Qp + (size_t)qbase * HDIM + 32, HDIM, lane);
    const _Float16 sc = (_Float16)0.125f;            // 1/sqrt(64), exact in f16
    #pragma unroll
    for (int i = 0; i < 16; ++i) { qf0[i] *= sc; qf1[i] *= sc; }

    v8f zero = {};
    v8f o[4] = {zero, zero, zero, zero};             // 16x64 context accumulator
    float mrow[8], lrow[8];
    #pragma unroll
    for (int r = 0; r < 8; ++r) { mrow[r] = -1e30f; lrow[r] = 0.0f; }

    uint32_t koff[2] = { (uint32_t)(uintptr_t)&kbuf[0][0], (uint32_t)(uintptr_t)&kbuf[1][0] };
    uint32_t voff[2] = { (uint32_t)(uintptr_t)&vbuf[0][0], (uint32_t)(uintptr_t)&vbuf[1][0] };

    // Prime buffer 0: K tile is one contiguous 4KB run (32 rows x 64 d);
    // V tile is 64 rows x 32 elems with row stride T_LEN.
    int cur = 0;
    tdm_load_tile_f16(koff[0], Kp, /*d0*/ 32 * HDIM, /*d1*/ 1, 32 * HDIM);
    tdm_load_tile_f16(voff[0], Vp, /*d0*/ 32,        /*d1*/ HDIM, T_LEN);
    __builtin_amdgcn_s_wait_tensorcnt(0);

    for (int jb = 0; jb < qlimit; jb += 32) {        // key blocks of 32
        int pre = (jb + 32) < qlimit;
        if (pre) {                                   // TDM prefetch next block
            tdm_load_tile_f16(koff[cur ^ 1], Kp + (size_t)(jb + 32) * HDIM,
                              32 * HDIM, 1, 32 * HDIM);
            tdm_load_tile_f16(voff[cur ^ 1], Vp + (jb + 32),
                              32, HDIM, T_LEN);
        }
        const h16* kl = &kbuf[cur][0];
        const h16* vl = &vbuf[cur][0];

        v8f s0 = zero, s1 = zero;
        v16h kf;
        kf = frag_b_nk(kl,                HDIM, lane);
        s0 = __builtin_amdgcn_wmma_f32_16x16x32_f16(false, qf0, false, kf, (short)0, s0, false, false);
        kf = frag_b_nk(kl + 32,           HDIM, lane);
        s0 = __builtin_amdgcn_wmma_f32_16x16x32_f16(false, qf1, false, kf, (short)0, s0, false, false);
        kf = frag_b_nk(kl + 16 * HDIM,      HDIM, lane);
        s1 = __builtin_amdgcn_wmma_f32_16x16x32_f16(false, qf0, false, kf, (short)0, s1, false, false);
        kf = frag_b_nk(kl + 16 * HDIM + 32, HDIM, lane);
        s1 = __builtin_amdgcn_wmma_f32_16x16x32_f16(false, qf1, false, kf, (short)0, s1, false, false);

        // online softmax over the 32 key columns (C layout: row (g*8+r), col nl)
        #pragma unroll
        for (int r = 0; r < 8; ++r) {
            int row = qbase + (g << 3) + r;
            int c0 = jb + nl, c1 = c0 + 16;
            float v0 = (c0 <= row) ? s0[r] : -1e30f; // causal mask
            float v1 = (c1 <= row) ? s1[r] : -1e30f;
            float bm = fmaxf(v0, v1);
            #pragma unroll
            for (int off = 8; off; off >>= 1) bm = fmaxf(bm, __shfl_xor(bm, off, 32));
            float mnew  = fmaxf(mrow[r], bm);
            float alpha = __expf(mrow[r] - mnew);
            mrow[r] = mnew;
            float p0 = __expf(v0 - mnew);
            float p1 = __expf(v1 - mnew);
            float ps = p0 + p1;
            #pragma unroll
            for (int off = 8; off; off >>= 1) ps += __shfl_xor(ps, off, 32);
            lrow[r] = lrow[r] * alpha + ps;
            o[0][r] *= alpha; o[1][r] *= alpha; o[2][r] *= alpha; o[3][r] *= alpha;
            int prow = (g << 3) + r;                 // C-layout -> row-major LDS
            pbuf[prow * 32 + nl]      = (h16)p0;
            pbuf[prow * 32 + nl + 16] = (h16)p1;
        }
        asm volatile("s_wait_dscnt 0" ::: "memory"); // P stores visible to A-frag loads

        v16h pa = frag_a_rowmajor(pbuf, 32, lane);   // relayout P as A fragment
        #pragma unroll
        for (int nt = 0; nt < 4; ++nt) {
            v16h vf = frag_b_nk(vl + (size_t)(nt * 16) * 32, 32, lane);
            o[nt] = __builtin_amdgcn_wmma_f32_16x16x32_f16(false, pa, false, vf,
                                                           (short)0, o[nt], false, false);
        }
        if (pre) {                                   // next tiles landed?
            __builtin_amdgcn_s_wait_tensorcnt(0);
            cur ^= 1;
        }
    }

    #pragma unroll
    for (int r = 0; r < 8; ++r) {
        float inv = 1.0f / lrow[r];
        int trow = qbase + (g << 3) + r;
        size_t base = ((size_t)(b * T_LEN + trow)) * D_MODEL + h * HDIM + nl;
        #pragma unroll
        for (int nt = 0; nt < 4; ++nt)
            ctx[base + nt * 16] = (h16)(o[nt][r] * inv);
    }
}

// ---------------------------------------------------------------------------
// Stage 3: output projection + bias, fp32 result. Same 32x64-per-wave tiling.
// ---------------------------------------------------------------------------
__global__ __launch_bounds__(256) void gemm_out_kernel(const h16* __restrict__ X,
                                                       const h16* __restrict__ Wt,
                                                       const float* __restrict__ bias,
                                                       float* __restrict__ out) {
    int lane = threadIdx.x & 31, wid = threadIdx.x >> 5;
    int m0 = blockIdx.y * 256 + wid * 32;
    int n0 = blockIdx.x * 64;
    v8f zero = {};
    v8f acc[2][4] = {{zero, zero, zero, zero}, {zero, zero, zero, zero}};
    for (int kk = 0; kk < D_MODEL; kk += 32) {
        v16h a0 = frag_a_rowmajor(X + (size_t)m0 * D_MODEL + kk, D_MODEL, lane);
        v16h a1 = frag_a_rowmajor(X + (size_t)(m0 + 16) * D_MODEL + kk, D_MODEL, lane);
        #pragma unroll
        for (int nt = 0; nt < 4; ++nt) {
            v16h bm = frag_b_nk(Wt + (size_t)(n0 + nt * 16) * D_MODEL + kk, D_MODEL, lane);
            acc[0][nt] = __builtin_amdgcn_wmma_f32_16x16x32_f16(false, a0, false, bm,
                                                                (short)0, acc[0][nt], false, false);
            acc[1][nt] = __builtin_amdgcn_wmma_f32_16x16x32_f16(false, a1, false, bm,
                                                                (short)0, acc[1][nt], false, false);
        }
    }
    int g = lane >> 4, nl = lane & 15;
    #pragma unroll
    for (int s = 0; s < 2; ++s) {
        #pragma unroll
        for (int nt = 0; nt < 4; ++nt) {
            int col = n0 + nt * 16 + nl;
            float bv = bias[col];
            #pragma unroll
            for (int r = 0; r < 8; ++r) {
                int rowg = m0 + s * 16 + (g << 3) + r;
                out[(size_t)rowg * D_MODEL + col] = acc[s][nt][r] + bv;
            }
        }
    }
}

// ---------------------------------------------------------------------------
extern "C" void kernel_launch(void* const* d_in, const int* in_sizes, int n_in,
                              void* d_out, int out_size, void* d_ws, size_t ws_size,
                              hipStream_t stream) {
    (void)in_sizes; (void)n_in; (void)out_size; (void)ws_size;
    const float* x  = (const float*)d_in[0];
    const float* Wq = (const float*)d_in[1];
    const float* Wk = (const float*)d_in[2];
    const float* Wv = (const float*)d_in[3];
    const float* Wo = (const float*)d_in[4];
    const float* bo = (const float*)d_in[5];
    float* out = (float*)d_out;

    char* ws = (char*)d_ws;
    size_t off = 0;
    auto take = [&](size_t bytes) -> char* {
        char* p = ws + off;
        off += (bytes + 255) & ~(size_t)255;
        return p;
    };
    const size_t XH_B  = (size_t)M_TOT * D_MODEL * sizeof(h16);      // 8 MB
    const size_t W_B   = (size_t)D_MODEL * D_MODEL * sizeof(h16);    // 2 MB each
    const size_t QKV_B = (size_t)BATCH * NHEAD * T_LEN * HDIM * sizeof(h16); // 8 MB each

    h16* xh  = (h16*)take(XH_B);
    h16* wqt = (h16*)take(W_B);
    h16* wkt = (h16*)take(W_B);
    h16* wvt = (h16*)take(W_B);
    h16* wot = (h16*)take(W_B);
    h16* qb  = (h16*)take(QKV_B);
    h16* kb  = (h16*)take(QKV_B);
    h16* vb  = (h16*)take(QKV_B);
    h16* ctx = (h16*)take(XH_B);        // total ~48 MB, L2-resident

    // Stage 0: casts
    int nx = M_TOT * D_MODEL;
    cast_f16_kernel<<<(nx + 255) / 256, 256, 0, stream>>>(x, xh, nx);
    transpose_cast_kernel<<<4096, 256, 0, stream>>>(Wq, wqt);
    transpose_cast_kernel<<<4096, 256, 0, stream>>>(Wk, wkt);
    transpose_cast_kernel<<<4096, 256, 0, stream>>>(Wv, wvt);
    transpose_cast_kernel<<<4096, 256, 0, stream>>>(Wo, wot);

    // Stage 1: QKV projections (V stored transposed for P*V B-fragments)
    dim3 gg(D_MODEL / 64, M_TOT / 256);   // (16, 16)
    gemm_qkv_kernel<<<gg, 256, 0, stream>>>(xh, wqt, qb, 0);
    gemm_qkv_kernel<<<gg, 256, 0, stream>>>(xh, wkt, kb, 0);
    gemm_qkv_kernel<<<gg, 256, 0, stream>>>(xh, wvt, vb, 1);

    // Stage 2: causal flash attention (1 wave per workgroup, TDM-fed)
    attn_fa_kernel<<<4096, 32, 0, stream>>>(qb, kb, vb, ctx);

    // Stage 3: output projection + bias -> fp32
    gemm_out_kernel<<<gg, 256, 0, stream>>>(ctx, wot, bo, out);
}
